// SIMCLR_31215822307944
// MI455X (gfx1250) — matrix operand order repaired and checked by reading, compiled, not run on hardware
//
#include <hip/hip_runtime.h>
#include <hip/hip_fp16.h>

typedef __attribute__((ext_vector_type(16))) _Float16 v16h;
typedef __attribute__((ext_vector_type(8)))  _Float16 v8h;
typedef __attribute__((ext_vector_type(8)))  float    v8f;

// Problem sizes (fixed by reference)
#define NROW   4096
#define FDIM   512
#define PDIM   128
#define M2     8192          // 2N rows
#define NSPLIT 8             // column splits of sim kernel

// ---------------- workspace byte offsets ----------------
#define XF_OFF   0u                         // 8192*512 f16   = 8 MiB
#define WP_OFF   8388608u                   // 65536 f16      = 128 KiB
#define H_OFF    8519680u                   // 8192*128 f32   = 4 MiB
#define PS_OFF   12713984u                  // 128*128 f32    = 64 KiB
#define PQ_OFF   12779520u                  // 128*128 f32    = 64 KiB
#define MUR_OFF  12845056u                  // 2*2*128 f32    = 2 KiB
#define ZN_OFF   12847104u                  // 8192*128 f16   = 2 MiB
#define RSP_OFF  14944256u                  // 8*8192 f32     = 256 KiB
#define POS_OFF  15206400u                  // 8192 f32       = 32 KiB

__device__ inline void wait_async0() {
#if __has_builtin(__builtin_amdgcn_s_wait_asynccnt)
  __builtin_amdgcn_s_wait_asynccnt(0);
#else
  asm volatile("s_wait_asynccnt 0x0" ::: "memory");
#endif
}

// async global -> LDS, 16 bytes per lane (ASYNCcnt-tracked, no VGPR round-trip)
__device__ inline void async_copy_b128(unsigned lds_byte_addr, const void* gaddr) {
  asm volatile("global_load_async_to_lds_b128 %0, %1, off"
               :: "v"(lds_byte_addr), "v"((unsigned long long)gaddr)
               : "memory");
}

// A fragment for v_wmma_f32_16x16x32_f16 from row-major f16 matrix.
// lane: m = lane&15, half = lane>>4.
// halfs 0..7  = Z[row][kbase + half*16 + 0..7]
// halfs 8..15 = Z[row][kbase + half*16 + 8 + 0..7]
__device__ inline v16h load_a_frag(const _Float16* __restrict__ Z, int ld,
                                   int row, int kbase, int half) {
  union { v16h v; v8h h[2]; } u;
  const _Float16* p = Z + (size_t)row * ld + kbase + half * 16;
  u.h[0] = *(const v8h*)p;
  u.h[1] = *(const v8h*)(p + 8);
  return u.v;
}

// ---------------- kernel 0: cast [x;xt] -> f16, 8 elems/thread ----------------
__global__ void simclr_pack_x(const float* __restrict__ x,
                              const float* __restrict__ xt,
                              _Float16* __restrict__ xf) {
  size_t base = (size_t)(blockIdx.x * blockDim.x + threadIdx.x) * 8;
  const size_t half0 = (size_t)NROW * FDIM;
  const float* src = (base < half0) ? (x + base) : (xt + (base - half0));
  v8h o;
#pragma unroll
  for (int i = 0; i < 8; ++i) o[i] = (_Float16)src[i];
  *(v8h*)(xf + base) = o;
}

// ---------------- kernel 1: pack W [512,128] into B-fragment order ----------------
// wp[nt*8192 + kt*512 + lane*16 + e] = W[kt*32 + (lane>>4)*16 + e][nt*16 + (lane&15)]
__global__ void simclr_pack_w(const float* __restrict__ W, _Float16* __restrict__ wp) {
  int idx  = blockIdx.x * blockDim.x + threadIdx.x;   // 0..65535
  int e    = idx & 15;
  int lane = (idx >> 4) & 31;
  int kt   = (idx >> 9) & 15;
  int nt   = idx >> 13;
  int n = nt * 16 + (lane & 15);
  int k = kt * 32 + ((lane >> 4) * 16) + e;
  wp[idx] = (_Float16)W[(size_t)k * PDIM + n];
}

// ---------------- kernel 2: projection h = [x;xt] @ W + b (WMMA f16) ----------------
// Two accumulators over even/odd k-tiles so consecutive v_wmma are independent
// (no D->C hazard nops); summed in the epilogue.
__global__ void __launch_bounds__(256)
simclr_proj(const _Float16* __restrict__ xf, const _Float16* __restrict__ wp,
            const float* __restrict__ bias, float* __restrict__ h) {
  int rt   = blockIdx.x;           // 0..511 row tile
  int wave = threadIdx.x >> 5;     // = nt, 8 col tiles cover all 128 cols
  int lane = threadIdx.x & 31;
  int half = lane >> 4;
  int row  = rt * 16 + (lane & 15);
  v8f acc0 = {}, acc1 = {};
#pragma unroll 2
  for (int kt = 0; kt < 16; kt += 2) {
    v16h ae = load_a_frag(xf, FDIM, row, kt * 32, half);
    v16h ao = load_a_frag(xf, FDIM, row, (kt + 1) * 32, half);
    union { v16h v; v8h h2[2]; } be, bo;
    const _Float16* pe = wp + ((size_t)(wave * 16 + kt) * 32 + lane) * 16;
    be.h2[0] = *(const v8h*)pe;
    be.h2[1] = *(const v8h*)(pe + 8);
    const _Float16* po = pe + 512;   // next k-tile: +32 lanes * 16 halfs
    bo.h2[0] = *(const v8h*)po;
    bo.h2[1] = *(const v8h*)(po + 8);
    acc0 = __builtin_amdgcn_wmma_f32_16x16x32_f16(false, ae, false, be.v,
                                                  (short)0, acc0, false, false);
    acc1 = __builtin_amdgcn_wmma_f32_16x16x32_f16(false, ao, false, bo.v,
                                                  (short)0, acc1, false, false);
  }
  int gcol = wave * 16 + (lane & 15);
  float bv = bias[gcol];
#pragma unroll
  for (int r = 0; r < 8; ++r) {
    int grow = rt * 16 + r + 8 * half;
    h[(size_t)grow * PDIM + gcol] = acc0[r] + acc1[r] + bv;
  }
}

// ---------------- kernel 3: per-view column partial sums ----------------
__global__ void simclr_stats_part(const float* __restrict__ h,
                                  float* __restrict__ psum, float* __restrict__ psq) {
  int wg    = blockIdx.x;          // 0..63
  int view  = wg >> 5;
  int chunk = wg & 31;             // 128 rows per chunk
  int c     = threadIdx.x & 127;
  int rsub  = threadIdx.x >> 7;
  int row0  = view * NROW + chunk * 128 + rsub;
  float s = 0.f, sq = 0.f;
  for (int i = 0; i < 64; ++i) {
    float v = h[(size_t)(row0 + 2 * i) * PDIM + c];
    s += v; sq += v * v;
  }
  int slot = wg * 2 + rsub;        // view slots: [view*64, view*64+64)
  psum[slot * PDIM + c] = s;
  psq [slot * PDIM + c] = sq;
}

// ---------------- kernel 4: finalize mu / rstd per view/col ----------------
__global__ void simclr_stats_final(const float* __restrict__ psum,
                                   const float* __restrict__ psq,
                                   float* __restrict__ mur) {
  int t = threadIdx.x;             // 256 = 2 views * 128 cols
  int view = t >> 7, c = t & 127;
  float s = 0.f, sq = 0.f;
  for (int slot = view * 64; slot < view * 64 + 64; ++slot) {
    s  += psum[slot * PDIM + c];
    sq += psq [slot * PDIM + c];
  }
  float mu  = s * (1.0f / NROW);
  float var = sq * (1.0f / NROW) - mu * mu;
  mur[view * 256 + c]       = mu;
  mur[view * 256 + 128 + c] = rsqrtf(var + 1e-5f);
}

// ---------------- kernel 5: BN + cosine row-normalize -> zn (f16) ----------------
__global__ void simclr_normalize(const float* __restrict__ h, const float* __restrict__ mur,
                                 const float* __restrict__ gamma, const float* __restrict__ beta,
                                 _Float16* __restrict__ zn) {
  __shared__ float red[256];
  int t = threadIdx.x;
  int rsub = t >> 7, c = t & 127;
  int row  = blockIdx.x * 2 + rsub;
  int view = row >> 12;
  float hv = h[(size_t)row * PDIM + c];
  float y  = gamma[c] * (hv - mur[view * 256 + c]) * mur[view * 256 + 128 + c] + beta[c];
  red[t] = y * y;
  __syncthreads();
  for (int s = 64; s > 0; s >>= 1) {
    if (c < s) red[t] += red[t + s];
    __syncthreads();
  }
  float norm = fmaxf(sqrtf(red[rsub * 128]), 1e-8f);
  zn[(size_t)row * PDIM + c] = (_Float16)(y / norm);
}

// per-tile epilogue: exp, diagonal exclusion, positive-pair capture, row partials
__device__ inline void tile_epilogue(const v8f& c, int ct, int diag_tile, int pos_tile,
                                     int rd, int rowbase, int n,
                                     float* rsum, float* __restrict__ pos) {
  float e[8];
#pragma unroll
  for (int r = 0; r < 8; ++r) e[r] = __expf(c[r] * 10.0f);   // 1/TEMP = 10
  if (ct == diag_tile) {              // zero main diagonal
#pragma unroll
    for (int r = 0; r < 8; ++r) if (r != rd) rsum[r] += e[r];
  } else {
#pragma unroll
    for (int r = 0; r < 8; ++r) rsum[r] += e[r];
    if (ct == pos_tile) {             // positive pair: diagonal of this tile
      float pv = e[0];
#pragma unroll
      for (int r = 1; r < 8; ++r) pv = (rd == r) ? e[r] : pv;
      if (rd >= 0 && rd < 8) pos[rowbase + n] = pv;
    }
  }
}

// ---------------- kernel 6: fused sim GEMM + exp + rowsum + diagonals ----------------
// Grid: 64 row-blocks (128 rows, 8 waves x 16-row strips) x 8 column splits.
// Per iteration: TWO 16x16 column tiles staged to LDS via async b128 copies
// (shared by all 8 waves), two independent accumulators so consecutive
// v_wmma ops have no D->src hazard, halved barrier rate.
__global__ void __launch_bounds__(256)
simclr_sim(const _Float16* __restrict__ zn,
           float* __restrict__ rsp,     // [NSPLIT][8192] partial row sums
           float* __restrict__ pos) {   // [8192]
  __shared__ _Float16 bt[4096];         // 8 KiB: 2 tiles x [kt][lane][16 halfs]
  int rb   = blockIdx.x >> 3;           // 0..63
  int cs   = blockIdx.x & 7;            // column split
  int wave = threadIdx.x >> 5;
  int lane = threadIdx.x & 31;
  int half = lane >> 4;
  int n    = lane & 15;
  int rt      = rb * 8 + wave;          // this wave's row tile (0..511)
  int rowbase = rt * 16;

  // A fragments for K = 0..127 (4 k-tiles), held for the whole kernel
  v16h a0 = load_a_frag(zn, PDIM, rowbase + n, 0,  half);
  v16h a1 = load_a_frag(zn, PDIM, rowbase + n, 32, half);
  v16h a2 = load_a_frag(zn, PDIM, rowbase + n, 64, half);
  v16h a3 = load_a_frag(zn, PDIM, rowbase + n, 96, half);

  float rsum[8];
#pragma unroll
  for (int r = 0; r < 8; ++r) rsum[r] = 0.f;

  const int diag_tile = rt;
  const int pos_tile  = (rt + 256) & 511;
  const int rd        = n - 8 * half;   // diag/pos register index if 0<=rd<8

  // LDS staging map: thread t moves 16 bytes of each of the two B tiles
  int slane = (threadIdx.x >> 1) & 31;
  int srow  = slane & 15;
  int scol  = (threadIdx.x >> 6) * 32 + (slane >> 4) * 16 + (threadIdx.x & 1) * 8;
  unsigned ldsbase = (unsigned)(size_t)(void*)bt;        // LDS offset in low 32 bits
  unsigned lds_t0  = ldsbase + (unsigned)threadIdx.x * 16u;
  unsigned lds_t1  = lds_t0 + 4096u;

  int ct0 = cs * 64;
  for (int ctp = 0; ctp < 32; ++ctp) {
    int ct = ct0 + 2 * ctp;
    __syncthreads();                    // previous readers done
    async_copy_b128(lds_t0, zn + (size_t)(ct * 16 + srow) * PDIM + scol);
    async_copy_b128(lds_t1, zn + (size_t)((ct + 1) * 16 + srow) * PDIM + scol);
    wait_async0();                      // this wave's LDS writes landed
    __syncthreads();                    // all waves' writes landed

    v8f c0 = {}, c1 = {};
#pragma unroll
    for (int kt = 0; kt < 4; ++kt) {
      union { v16h v; v8h h2[2]; } b0, b1;
      const _Float16* p0 = bt + (kt * 32 + lane) * 16;
      b0.h2[0] = *(const v8h*)p0;
      b0.h2[1] = *(const v8h*)(p0 + 8);
      const _Float16* p1 = p0 + 2048;
      b1.h2[0] = *(const v8h*)p1;
      b1.h2[1] = *(const v8h*)(p1 + 8);
      v16h a = (kt == 0) ? a0 : (kt == 1) ? a1 : (kt == 2) ? a2 : a3;
      c0 = __builtin_amdgcn_wmma_f32_16x16x32_f16(false, a, false, b0.v,
                                                  (short)0, c0, false, false);
      c1 = __builtin_amdgcn_wmma_f32_16x16x32_f16(false, a, false, b1.v,
                                                  (short)0, c1, false, false);
    }

    tile_epilogue(c0, ct,     diag_tile, pos_tile, rd, rowbase, n, rsum, pos);
    tile_epilogue(c1, ct + 1, diag_tile, pos_tile, rd, rowbase, n, rsum, pos);
  }

  // reduce partial row sums across the 16-lane halves, write split partial
#pragma unroll
  for (int r = 0; r < 8; ++r) {
    float v = rsum[r];
    v += __shfl_xor(v, 1, 32);
    v += __shfl_xor(v, 2, 32);
    v += __shfl_xor(v, 4, 32);
    v += __shfl_xor(v, 8, 32);
    if (n == r) rsp[(size_t)cs * M2 + rowbase + 8 * half + r] = v;
  }
}

// ---------------- kernel 7: V = -log(pos / rowsum) ----------------
__global__ void simclr_final(const float* __restrict__ rsp, const float* __restrict__ pos,
                             float* __restrict__ out) {
  int i = blockIdx.x * 256 + threadIdx.x;
  float s = 0.f;
#pragma unroll
  for (int cs = 0; cs < NSPLIT; ++cs) s += rsp[(size_t)cs * M2 + i];
  out[i] = -__logf(pos[i] / s);
}

extern "C" void kernel_launch(void* const* d_in, const int* in_sizes, int n_in,
                              void* d_out, int out_size, void* d_ws, size_t ws_size,
                              hipStream_t stream) {
  const float* x     = (const float*)d_in[0];
  const float* xt    = (const float*)d_in[1];
  const float* W     = (const float*)d_in[2];
  const float* b     = (const float*)d_in[3];
  const float* gamma = (const float*)d_in[4];
  const float* beta  = (const float*)d_in[5];
  float* out = (float*)d_out;

  char* ws = (char*)d_ws;
  _Float16* xf  = (_Float16*)(ws + XF_OFF);
  _Float16* wp  = (_Float16*)(ws + WP_OFF);
  float*    h   = (float*)   (ws + H_OFF);
  float*    ps  = (float*)   (ws + PS_OFF);
  float*    pq  = (float*)   (ws + PQ_OFF);
  float*    mur = (float*)   (ws + MUR_OFF);
  _Float16* zn  = (_Float16*)(ws + ZN_OFF);
  float*    rsp = (float*)   (ws + RSP_OFF);
  float*    pos = (float*)   (ws + POS_OFF);

  simclr_pack_x     <<<2048, 256, 0, stream>>>(x, xt, xf);
  simclr_pack_w     <<<256,  256, 0, stream>>>(W, wp);
  simclr_proj       <<<512,  256, 0, stream>>>(xf, wp, b, h);
  simclr_stats_part <<<64,   256, 0, stream>>>(h, ps, pq);
  simclr_stats_final<<<1,    256, 0, stream>>>(ps, pq, mur);
  simclr_normalize  <<<4096, 256, 0, stream>>>(h, mur, gamma, beta, zn);
  simclr_sim        <<<512,  256, 0, stream>>>(zn, rsp, pos);
  simclr_final      <<<32,   256, 0, stream>>>(rsp, pos, out);
}
